// MultiHeadSelfAttention_54924041781256
// MI455X (gfx1250) — compile-verified, matrix operand order checked by reading, not run on hardware
//
#include <hip/hip_runtime.h>

#define B_  2
#define S_  2048
#define D_  1024
#define H_  16
#define DK_ 64

typedef __attribute__((ext_vector_type(16))) __bf16 v16bf;
typedef __attribute__((ext_vector_type(8)))  float  v8f;
typedef __attribute__((ext_vector_type(4)))  unsigned int u32x4;
typedef __attribute__((ext_vector_type(4)))  int i32x4;
typedef __attribute__((ext_vector_type(8)))  int i32x8;

#if __has_include(<hip/amd_detail/amd_gfx1250_TDM.h>)
#define TDM_6ARG 1
#else
#define TDM_6ARG 0
#endif

union BF16Frag {
  v16bf v;
  uint4 q[2];
  unsigned short u[16];
};

// round-to-nearest-even f32 -> bf16 bits
__device__ __forceinline__ unsigned short f2bf(float f) {
  unsigned int u = __float_as_uint(f);
  unsigned int r = u + 0x7FFFu + ((u >> 16) & 1u);
  return (unsigned short)(r >> 16);
}

// Per-lane WMMA A/B operand from a contiguous bf16 row: two 16B chunks at +0,+32B.
__device__ __forceinline__ v16bf load_row_frag(const unsigned short* rowk) {
  BF16Frag f;
  f.q[0] = *(const uint4*)(rowk);
  f.q[1] = *(const uint4*)(rowk + 16);
  return f.v;
}

__device__ __forceinline__ v8f wmma_bf16(v16bf a, v16bf b, v8f c) {
  return __builtin_amdgcn_wmma_f32_16x16x32_bf16(false, a, false, b, (short)0, c,
                                                 false, false);
}

// ---- Tensor Data Mover: 2D tile (bf16) global -> LDS ----
// D# bitfields per CDNA5 ISA.
__device__ __forceinline__ void tdm_load_2d_bf16(
    unsigned int lds_off_bytes, const void* gaddr,
    unsigned int tile_d0, unsigned int tile_d1,
    unsigned int tens_d0, unsigned int tens_d1, unsigned int d0_stride) {
  unsigned long long ga = (unsigned long long)(size_t)gaddr;
  u32x4 g0;
  g0[0] = 1u;                                      // count=1, is_restore=0
  g0[1] = lds_off_bytes;                           // lds_addr
  g0[2] = (unsigned int)(ga & 0xFFFFFFFFu);        // global_addr[31:0]
  g0[3] = (unsigned int)((ga >> 32) & 0x01FFFFFFu) | (2u << 30);  // addr[56:32] | type=2
  i32x8 g1;
  g1[0] = (int)(1u << 16);                         // workgroup_mask=0, data_size=1 (2B)
  g1[1] = (int)(tens_d0 << 16);                    // tensor_dim0[15:0] -> bits 63:48
  g1[2] = (int)((tens_d0 >> 16) | (tens_d1 << 16));// tensor_dim0 hi | tensor_dim1 lo
  g1[3] = (int)((tens_d1 >> 16) | (tile_d0 << 16));// tensor_dim1 hi | tile_dim0
  g1[4] = (int)(tile_d1 & 0xFFFFu);                // tile_dim1 (tile_dim2=0)
  g1[5] = (int)d0_stride;                          // tensor_dim0_stride[31:0]
  g1[6] = 0;                                       // stride hi | dim1_stride lo
  g1[7] = 0;
  i32x4 z4 = {0, 0, 0, 0};
#if TDM_6ARG
  i32x8 z8 = {0, 0, 0, 0, 0, 0, 0, 0};
  __builtin_amdgcn_tensor_load_to_lds(g0, g1, z4, z4, z8, 0);
#else
  __builtin_amdgcn_tensor_load_to_lds(g0, g1, z4, z4, 0);
#endif
}

// ---------------- kernel 0: f32 -> bf16 convert ----------------
__global__ void k_convert(const float* __restrict__ src,
                          unsigned short* __restrict__ dst, int n) {
  int i = blockIdx.x * blockDim.x + threadIdx.x;
  int stride = gridDim.x * blockDim.x;
  for (; i < n; i += stride) dst[i] = f2bf(src[i]);
}

// ---- Shared GEMM core: block = 4 waves, computes rows [32mt,32mt+32) x 256 cols.
// Activation tile (32 x 1024 bf16) is TDM-staged into LDS, double-buffered as
// 2 x (32 x 512) halves; weights stream from global (L2-resident).
// Wave w owns cols [nblk*256 + 64w, +64): acc[2][4] = 32x64 in registers.
__device__ __forceinline__ void gemm_block_32x256(
    const unsigned short* __restrict__ A,     // [4096,1024] bf16 row-major
    const unsigned short* __restrict__ Wm,    // [1024,1024] bf16 row-major
    int mt, int nbase_wave, int l, int w, char* dynsmem, v8f acc[2][4]) {
  const int lr = l & 15, khalf = l >> 4;
  unsigned short* lbuf = (unsigned short*)dynsmem;
  const unsigned int lds_base = (unsigned int)(size_t)dynsmem;
  const unsigned short* arows = A + (size_t)(mt * 32) * D_;

  const unsigned short* brow[4];
#pragma unroll
  for (int ni = 0; ni < 4; ++ni)
    brow[ni] = Wm + (size_t)(nbase_wave + ni * 16 + lr) * D_ + khalf * 8;

  if (w == 0) tdm_load_2d_bf16(lds_base, arows, 512, 32, D_, B_ * S_, D_);

  for (int st = 0; st < 2; ++st) {
    if (w == 0) __builtin_amdgcn_s_wait_tensorcnt(0);
    __syncthreads();
    if (w == 0 && st == 0)
      tdm_load_2d_bf16(lds_base + 32 * 512 * 2, arows + 512, 512, 32, D_,
                       B_ * S_, D_);
    const unsigned short* abuf = lbuf + st * (32 * 512);
    const int soff = st * 512;
    for (int kc = 0; kc < 16; ++kc) {
      if (kc + 6 < 16) {
        __builtin_prefetch(brow[0] + soff + (kc + 6) * 32, 0, 1);
        __builtin_prefetch(brow[2] + soff + (kc + 6) * 32, 0, 1);
      }
      v16bf afr[2];
#pragma unroll
      for (int mi = 0; mi < 2; ++mi)
        afr[mi] =
            load_row_frag(abuf + (mi * 16 + lr) * 512 + kc * 32 + khalf * 8);
#pragma unroll
      for (int ni = 0; ni < 4; ++ni) {
        v16bf bf = load_row_frag(brow[ni] + soff + kc * 32);
        acc[0][ni] = wmma_bf16(afr[0], bf, acc[0][ni]);
        acc[1][ni] = wmma_bf16(afr[1], bf, acc[1][ni]);
      }
    }
    __syncthreads();
  }
}

// ---------------- kernel 1: fused QKV projection ----------------
// grid = 3 proj x 128 mtiles x 4 nblocks. Q,K -> [B,H,S,DK]; V -> [B,H,DK,S].
__global__ __launch_bounds__(128) void k_qkv(
    const unsigned short* __restrict__ xb,
    const unsigned short* __restrict__ wqb,
    const unsigned short* __restrict__ wkb,
    const unsigned short* __restrict__ wvb,
    const float* __restrict__ bq, const float* __restrict__ bk,
    const float* __restrict__ bv,
    unsigned short* __restrict__ Qb, unsigned short* __restrict__ Kb,
    unsigned short* __restrict__ Vt) {
  extern __shared__ char dynsmem[];
  const int l = threadIdx.x & 31;
  const int w = threadIdx.x >> 5;
  int proj = blockIdx.x / 512;
  int rem = blockIdx.x % 512;
  int mt = rem / 4;
  int nblk = rem % 4;
  int nbase = nblk * 256 + w * 64;

  const unsigned short* wsel = (proj == 0) ? wqb : (proj == 1) ? wkb : wvb;
  const float* bsel = (proj == 0) ? bq : (proj == 1) ? bk : bv;

  v8f acc[2][4] = {};
  gemm_block_32x256(xb, wsel, mt, nbase, l, w, dynsmem, acc);

  const int lr = l & 15, khalf = l >> 4;
  float bias[4];
  int oc[4], hh[4], dk[4];
#pragma unroll
  for (int ni = 0; ni < 4; ++ni) {
    oc[ni] = nbase + ni * 16 + lr;
    bias[ni] = bsel[oc[ni]];
    hh[ni] = oc[ni] >> 6;
    dk[ni] = oc[ni] & 63;
  }
#pragma unroll
  for (int mi = 0; mi < 2; ++mi) {
#pragma unroll
    for (int r = 0; r < 8; ++r) {
      int mg = mt * 32 + mi * 16 + r + 8 * khalf;
      int bb = mg >> 11, s = mg & (S_ - 1);
#pragma unroll
      for (int ni = 0; ni < 4; ++ni) {
        unsigned short val = f2bf(acc[mi][ni][r] + bias[ni]);
        if (proj == 2)
          Vt[(((size_t)bb * H_ + hh[ni]) * DK_ + dk[ni]) * S_ + s] = val;
        else if (proj == 0)
          Qb[(((size_t)bb * H_ + hh[ni]) * S_ + s) * DK_ + dk[ni]] = val;
        else
          Kb[(((size_t)bb * H_ + hh[ni]) * S_ + s) * DK_ + dk[ni]] = val;
      }
    }
  }
}

// ---------------- kernel 2: attention (scores -> softmax -> ctx) ----------------
// One block = (b, h, 16 query rows). 4 waves. Dynamic LDS: 16 x 2048 f32 = 128 KB.
__global__ __launch_bounds__(128) void k_attn(
    const unsigned short* __restrict__ Qb, const unsigned short* __restrict__ Kb,
    const unsigned short* __restrict__ Vt, float* __restrict__ attn_out,
    unsigned short* __restrict__ Ctx) {
  extern __shared__ char dynsmem[];
  float* sc = (float*)dynsmem;  // [16][S_]
  const int l = threadIdx.x & 31, w = threadIdx.x >> 5;
  const int lr = l & 15, khalf = l >> 4;
  int qb = blockIdx.x & 127;
  int h = (blockIdx.x >> 7) & 15;
  int bb = blockIdx.x >> 11;
  const size_t headQK = ((size_t)bb * H_ + h) * (size_t)S_ * DK_;

  // ---- Phase 1: scores; wave w owns key-tiles [32w, 32w+32) ----
  const unsigned short* qrow =
      Qb + headQK + (size_t)(qb * 16 + lr) * DK_ + khalf * 8;
  v16bf a0 = load_row_frag(qrow);
  v16bf a1 = load_row_frag(qrow + 32);
  for (int kt = w * 32; kt < w * 32 + 32; ++kt) {
    const unsigned short* krow =
        Kb + headQK + (size_t)(kt * 16 + lr) * DK_ + khalf * 8;
    v16bf b0 = load_row_frag(krow);
    v16bf b1 = load_row_frag(krow + 32);
    v8f c = {};
    c = wmma_bf16(a0, b0, c);
    c = wmma_bf16(a1, b1, c);
#pragma unroll
    for (int r = 0; r < 8; ++r)
      sc[(r + 8 * khalf) * S_ + kt * 16 + lr] = c[r] * 0.125f;  // 1/sqrt(64)
  }
  __syncthreads();

  // ---- Phase 2: softmax; wave w handles rows 4w..4w+3 ----
  float* attn_row_base =
      attn_out + (((size_t)bb * H_ + h) * S_ + (size_t)qb * 16) * S_;
  for (int rr = 0; rr < 4; ++rr) {
    int row = w * 4 + rr;
    float* srow = sc + row * S_;
    float mx = -3.4e38f;
    for (int k = l; k < S_; k += 32) mx = fmaxf(mx, srow[k]);
    for (int off = 16; off > 0; off >>= 1) mx = fmaxf(mx, __shfl_xor(mx, off, 32));
    float sum = 0.f;
    for (int k = l; k < S_; k += 32) {
      float e = __expf(srow[k] - mx);
      srow[k] = e;
      sum += e;
    }
    for (int off = 16; off > 0; off >>= 1) sum += __shfl_xor(sum, off, 32);
    float inv = 1.0f / sum;
    float* grow = attn_row_base + (size_t)row * S_;
    for (int k = l; k < S_; k += 32) {
      float p = srow[k] * inv;
      srow[k] = p;
      grow[k] = p;  // materialize attn_weights (f32)
    }
  }
  __syncthreads();

  // ---- Phase 3: ctx = P @ V; wave w owns dk-tile w ----
  const unsigned short* vbase = Vt + ((size_t)bb * H_ + h) * (size_t)DK_ * S_ +
                                (size_t)(w * 16 + lr) * S_ + khalf * 8;
  const float* prow = sc + lr * S_ + khalf * 8;
  v8f acc = {};
  for (int kc = 0; kc < S_ / 32; ++kc) {
    if (kc + 4 < S_ / 32) __builtin_prefetch(vbase + (kc + 4) * 32, 0, 1);
    const float* p0 = prow + kc * 32;
    float4 pa = *(const float4*)(p0);
    float4 pb = *(const float4*)(p0 + 4);
    float4 pc = *(const float4*)(p0 + 16);
    float4 pd = *(const float4*)(p0 + 20);
    BF16Frag fa;
    fa.u[0] = f2bf(pa.x); fa.u[1] = f2bf(pa.y); fa.u[2] = f2bf(pa.z); fa.u[3] = f2bf(pa.w);
    fa.u[4] = f2bf(pb.x); fa.u[5] = f2bf(pb.y); fa.u[6] = f2bf(pb.z); fa.u[7] = f2bf(pb.w);
    fa.u[8] = f2bf(pc.x); fa.u[9] = f2bf(pc.y); fa.u[10] = f2bf(pc.z); fa.u[11] = f2bf(pc.w);
    fa.u[12] = f2bf(pd.x); fa.u[13] = f2bf(pd.y); fa.u[14] = f2bf(pd.z); fa.u[15] = f2bf(pd.w);
    v16bf b = load_row_frag(vbase + kc * 32);
    acc = wmma_bf16(fa.v, b, acc);
  }
  int d = h * DK_ + w * 16 + lr;
#pragma unroll
  for (int r = 0; r < 8; ++r) {
    int s = qb * 16 + r + 8 * khalf;
    Ctx[((size_t)bb * S_ + s) * D_ + d] = f2bf(acc[r]);
  }
}

// ---------------- kernel 3: output projection ----------------
__global__ __launch_bounds__(128) void k_oproj(
    const unsigned short* __restrict__ ctx, const unsigned short* __restrict__ wob,
    const float* __restrict__ bo, float* __restrict__ out) {
  extern __shared__ char dynsmem[];
  const int l = threadIdx.x & 31, w = threadIdx.x >> 5;
  int mt = blockIdx.x / 4;
  int nblk = blockIdx.x % 4;
  int nbase = nblk * 256 + w * 64;

  v8f acc[2][4] = {};
  gemm_block_32x256(ctx, wob, mt, nbase, l, w, dynsmem, acc);

  const int lr = l & 15, khalf = l >> 4;
  float bias[4];
#pragma unroll
  for (int ni = 0; ni < 4; ++ni) bias[ni] = bo[nbase + ni * 16 + lr];
#pragma unroll
  for (int mi = 0; mi < 2; ++mi) {
#pragma unroll
    for (int r = 0; r < 8; ++r) {
      int mg = mt * 32 + mi * 16 + r + 8 * khalf;
#pragma unroll
      for (int ni = 0; ni < 4; ++ni)
        out[(size_t)mg * D_ + nbase + ni * 16 + lr] = acc[mi][ni][r] + bias[ni];
    }
  }
}

extern "C" void kernel_launch(void* const* d_in, const int* in_sizes, int n_in,
                              void* d_out, int out_size, void* d_ws, size_t ws_size,
                              hipStream_t stream) {
  (void)in_sizes; (void)n_in; (void)out_size; (void)ws_size;
  const float* x  = (const float*)d_in[0];
  const float* Wq = (const float*)d_in[1];
  const float* bq = (const float*)d_in[2];
  const float* Wk = (const float*)d_in[3];
  const float* bk = (const float*)d_in[4];
  const float* Wv = (const float*)d_in[5];
  const float* bv = (const float*)d_in[6];
  const float* Wo = (const float*)d_in[7];
  const float* bo = (const float*)d_in[8];
  float* out = (float*)d_out;

  const size_t XN = (size_t)B_ * S_ * D_;  // 4,194,304
  const size_t WN = (size_t)D_ * D_;       // 1,048,576

  unsigned short* ws  = (unsigned short*)d_ws;
  unsigned short* xb  = ws;
  unsigned short* wqb = xb + XN;
  unsigned short* wkb = wqb + WN;
  unsigned short* wvb = wkb + WN;
  unsigned short* wob = wvb + WN;
  unsigned short* Qb  = wob + WN;
  unsigned short* Kb  = Qb + XN;
  unsigned short* Vt  = Kb + XN;   // [B,H,DK,S]
  unsigned short* Ctx = Vt + XN;   // [B,S,D]

  k_convert<<<1024, 256, 0, stream>>>(x,  xb,  (int)XN);
  k_convert<<<512,  256, 0, stream>>>(Wq, wqb, (int)WN);
  k_convert<<<512,  256, 0, stream>>>(Wk, wkb, (int)WN);
  k_convert<<<512,  256, 0, stream>>>(Wv, wvb, (int)WN);
  k_convert<<<512,  256, 0, stream>>>(Wo, wob, (int)WN);

  // 3 proj x 128 mtiles x 4 nblocks; 64KB dynamic LDS (2x 32x512 bf16 buffers)
  k_qkv<<<1536, 128, 2 * 32 * 512 * 2, stream>>>(xb, wqb, wkb, wvb, bq, bk, bv,
                                                 Qb, Kb, Vt);

  float* attn_out = out + XN;  // outputs: [output | attn_weights]
  k_attn<<<4096, 128, 16 * 2048 * sizeof(float), stream>>>(Qb, Kb, Vt, attn_out,
                                                           Ctx);

  k_oproj<<<512, 128, 2 * 32 * 512 * 2, stream>>>(Ctx, wob, bo, out);
}
